// SelfAttention_11768210391275
// MI455X (gfx1250) — matrix operand order repaired and checked by reading, not compiled, not run on hardware
//
#include <hip/hip_runtime.h>
#include <hip/hip_bf16.h>
#include <cstdint>
#include <cstddef>

#define BB   4
#define CC   256
#define CQKD 32
#define NN   4096   // H*W = 64*64

typedef __attribute__((ext_vector_type(16))) __bf16 v16bf;
typedef __attribute__((ext_vector_type(8)))  float  v8f;
typedef __attribute__((ext_vector_type(4)))  unsigned int u32x4;
typedef __attribute__((ext_vector_type(8)))  int  i32x8;
typedef __attribute__((ext_vector_type(4)))  int  i32x4;

#if defined(__AMDGCN__) && __has_builtin(__builtin_amdgcn_tensor_load_to_lds)
#define ATTN_USE_TDM 1
#else
#define ATTN_USE_TDM 0
#endif

union BF16V {
    v16bf v;
    unsigned short u[16];
    uint4 q[2];
};

__device__ __forceinline__ unsigned short f2bf(float f) {
    union { float f; unsigned int u; } a;
    a.f = f;
    unsigned int r = a.u + 0x7FFFu + ((a.u >> 16) & 1u);  // round-to-nearest-even
    return (unsigned short)(r >> 16);
}

// ---------------------------------------------------------------------------
// Kernel 1: 1x1-conv projections q,k,v -> bf16 in workspace.
// Grid: BB * (NN/64) blocks, 256 threads. LDS tile: x[256 c][64 n] f32 (64KB).
// ---------------------------------------------------------------------------
__global__ __launch_bounds__(256) void proj_kernel(
    const float* __restrict__ x,
    const float* __restrict__ Wq, const float* __restrict__ bq,
    const float* __restrict__ Wk, const float* __restrict__ bk,
    const float* __restrict__ Wv, const float* __restrict__ bv,
    unsigned short* __restrict__ qb,
    unsigned short* __restrict__ kb,
    unsigned short* __restrict__ vb)
{
    __shared__ __align__(16) float xs[CC * 64];

    const int b  = blockIdx.x / (NN / 64);
    const int n0 = (blockIdx.x % (NN / 64)) * 64;
    const int t  = threadIdx.x;

    // cooperative load: thread t loads x row c=t (64 floats, b128-vectorized)
    {
        const float4* src = (const float4*)(x + ((size_t)b * CC + t) * NN + n0);
        float4* dst = (float4*)(xs + t * 64);
#pragma unroll
        for (int i = 0; i < 16; ++i) dst[i] = src[i];
    }
    __syncthreads();

    // ---- V projection: thread t computes output row r=t over 64 columns ----
    {
        float acc[64];
        const float bias = bv[t];
#pragma unroll
        for (int n = 0; n < 64; ++n) acc[n] = bias;
        const float* wrow = Wv + (size_t)t * CC;
        for (int c = 0; c < CC; ++c) {
            const float w = wrow[c];
            const float4* row = (const float4*)(xs + c * 64);
#pragma unroll
            for (int i = 0; i < 16; ++i) {
                float4 xv = row[i];
                acc[4*i+0] = fmaf(w, xv.x, acc[4*i+0]);
                acc[4*i+1] = fmaf(w, xv.y, acc[4*i+1]);
                acc[4*i+2] = fmaf(w, xv.z, acc[4*i+2]);
                acc[4*i+3] = fmaf(w, xv.w, acc[4*i+3]);
            }
        }
        unsigned short* dst = vb + ((size_t)b * CC + t) * NN + n0;
#pragma unroll
        for (int n = 0; n < 64; ++n) dst[n] = f2bf(acc[n]);
    }

    // ---- Q / K projections: threads 0..63 (r = t&31, q if t<32 else k) ----
    if (t < 64) {
        const int  r   = t & 31;
        const bool isq = (t < 32);
        const float* wrow = (isq ? Wq : Wk) + (size_t)r * CC;
        const float bias  = isq ? bq[r] : bk[r];
        float acc[64];
#pragma unroll
        for (int n = 0; n < 64; ++n) acc[n] = bias;
        for (int c = 0; c < CC; ++c) {
            const float w = wrow[c];
            const float4* row = (const float4*)(xs + c * 64);
#pragma unroll
            for (int i = 0; i < 16; ++i) {
                float4 xv = row[i];
                acc[4*i+0] = fmaf(w, xv.x, acc[4*i+0]);
                acc[4*i+1] = fmaf(w, xv.y, acc[4*i+1]);
                acc[4*i+2] = fmaf(w, xv.z, acc[4*i+2]);
                acc[4*i+3] = fmaf(w, xv.w, acc[4*i+3]);
            }
        }
        unsigned short* dst = (isq ? qb : kb) + ((size_t)b * CQKD + r) * NN + n0;
#pragma unroll
        for (int n = 0; n < 64; ++n) dst[n] = f2bf(acc[n]);
    }
}

// ---------------------------------------------------------------------------
// Kernel 2: flash-attention over N=4096, d=32, dv=256, all matmuls via
// v_wmma_f32_16x16x32_bf16.  V tiles are streamed into LDS by the Tensor
// Data Mover (tensor_load_to_lds + s_wait_tensorcnt), overlapping the DMA
// with the S-matmul + online-softmax phase.
// Grid: BB*(NN/64); 256 threads = 8 waves.
// Wave w: row-group rg=w>>1 (16 rows), c-half ch=w&1 (128 channels).
// ---------------------------------------------------------------------------
__global__ __launch_bounds__(256) void attn_kernel(
    const unsigned short* __restrict__ qb,
    const unsigned short* __restrict__ kb,
    const unsigned short* __restrict__ vb,
    const float* __restrict__ x,
    const float* __restrict__ gamma,
    float* __restrict__ out)
{
    __shared__ __align__(16) unsigned short q_lds[64 * 32];        // [j][d]  4KB
    __shared__ __align__(16) unsigned short v_lds[CC * 64];        // [c][n] 32KB
    __shared__ __align__(16) unsigned short p_lds[8 * 16 * 64];    // per-wave P 16KB

    const int b    = blockIdx.x / (NN / 64);
    const int i0   = (blockIdx.x % (NN / 64)) * 64;   // attention-row base
    const int tid  = threadIdx.x;
    const int wid  = tid >> 5;
    const int lane = tid & 31;
    const int rg   = wid >> 1;      // row group 0..3
    const int ch   = wid & 1;       // channel half 0..1
    const int lh   = lane >> 4;     // lane-half (ISA layout "hi")
    const int l16  = lane & 15;

    // --- preload A operand for S: rows of k (ISA 16-bit A 16x32 layout) ---
    // element e (VGPR e/2, half e%2):  K = e + 8*lh           (e < 8)
    //                                  K = 16 + (e-8) + 8*lh  (e >= 8)
    BF16V aQ;
    const int irow = i0 + 16 * rg + l16;
#pragma unroll
    for (int e = 0; e < 16; ++e) {
        const int K = (e < 8) ? (e + 8 * lh) : (16 + (e - 8) + 8 * lh);
        aQ.u[e] = kb[((size_t)b * CQKD + K) * NN + irow];
    }

#if ATTN_USE_TDM
    // Invariant part of the Tensor DMA Descriptor (D#), ISA ch.8 layout.
    // 2-D tile: tile_dim0 = 64 cols (bf16), tile_dim1 = 256 rows,
    // tensor_dim0 = NN, tensor_dim0_stride = NN, data_size = 2 bytes.
    i32x8 g1;
    g1[0] = 0x00010000;                      // wg_mask=0 | data_size(2B)<<16
    g1[1] = (int)((unsigned)NN << 16);       // tensor_dim0[15:0] @ bits 63:48
    g1[2] = (int)((unsigned)CC << 16);       // tensor_dim0[31:16]=0 | tensor_dim1[15:0]
    g1[3] = (int)(64u << 16);                // tensor_dim1[31:16]=0 | tile_dim0=64
    g1[4] = 256;                             // tile_dim1=256 | tile_dim2=0
    g1[5] = NN;                              // tensor_dim0_stride[31:0]
    g1[6] = 0;                               // stride0[47:32]=0 | dim1_stride lo=0
    g1[7] = 0;
    const i32x4 gz = {0, 0, 0, 0};           // groups 2/3 unused (2-D tensor)
    // ISA 10.2: flat addr low 32 bits map directly to the LDS byte offset.
    const unsigned lds_v = (unsigned)(uintptr_t)v_lds;
#endif

    v8f   O[8];
    float mrow[8], lrow[8];
#pragma unroll
    for (int t = 0; t < 8; ++t) O[t] = (v8f)0.0f;
#pragma unroll
    for (int r = 0; r < 8; ++r) { mrow[r] = -1.0e30f; lrow[r] = 0.0f; }

    unsigned short* pw = p_lds + wid * (16 * 64);   // wave-private P region

    for (int j0 = 0; j0 < NN; j0 += 64) {
        __syncthreads();   // previous tile fully consumed before overwrite

#if ATTN_USE_TDM
        // ---- wave 0 kicks the TDM for this iteration's V tile ----------
        // (uniform scalar branch: TDM ignores EXEC, so guard via control flow;
        //  descriptor values are wave-uniform -> SGPRs, as VIMAGE requires)
        if (wid == 0) {
            const unsigned long long ga =
                (unsigned long long)(uintptr_t)(vb + ((size_t)b * CC) * NN + j0);
            u32x4 g0;
            g0[0] = 1u;                                   // count=1, user mode
            g0[1] = lds_v;                                // lds_addr
            g0[2] = (unsigned)(ga & 0xFFFFFFFFu);         // global_addr[31:0]
            g0[3] = (unsigned)((ga >> 32) & 0x01FFFFFFu)  // global_addr[56:32]
                    | (2u << 30);                         // type=2 ("image")
#if __clang_major__ >= 23
            __builtin_amdgcn_tensor_load_to_lds(g0, g1, gz, gz,
                                                (i32x8)0, 0 /*cpol*/);
#else
            __builtin_amdgcn_tensor_load_to_lds(g0, g1, gz, gz, 0 /*cpol*/);
#endif
        }
#else
        // ---- fallback: cooperative load v tile -> v_lds[c][n] ----------
        {
            const uint4* src = (const uint4*)(vb + ((size_t)b * CC + tid) * NN + j0);
            uint4* dst = (uint4*)(v_lds + tid * 64);
#pragma unroll
            for (int i = 0; i < 8; ++i) dst[i] = src[i];
        }
#endif

        // ---- cooperative load q tile -> q_lds[j][K] (j=0..63, K=0..31) ----
        {
            const int j  = tid & 63;
            const int k8 = (tid >> 6) * 8;
            unsigned short tmp[8];
#pragma unroll
            for (int kk = 0; kk < 8; ++kk)
                tmp[kk] = qb[((size_t)b * CQKD + k8 + kk) * NN + j0 + j];
            uint4* dst = (uint4*)(q_lds + j * 32 + k8);
            dst[0] = *(const uint4*)tmp;
        }
        if (j0 + 64 < NN)   // hint next q tile into cache (global_prefetch_b8)
            __builtin_prefetch(qb + ((size_t)b * CQKD + (tid >> 6) * 8) * NN
                               + j0 + 64 + (tid & 63), 0, 3);
        __syncthreads();    // q ready (V arrives via TDM during S/softmax)

        // ---- S = (k rows) . (q cols): 4 tiles of 16x16, K=32 = full d ----
        v8f s[4];
#pragma unroll
        for (int t = 0; t < 4; ++t) {
            BF16V bQ;   // B layout: lane col j=l16, element e -> K = e + 8*lh (e<8), 16 + (e-8) + 8*lh (e>=8)
            const uint4* p = (const uint4*)(q_lds + (16 * t + l16) * 32 + 8 * lh);
            bQ.q[0] = p[0];
            bQ.q[1] = p[2];
            v8f z = (v8f)0.0f;
            s[t] = __builtin_amdgcn_wmma_f32_16x16x32_bf16(
                false, aQ.v, false, bQ.v, (short)0, z, false, false);
        }

        // ---- online softmax (rows M = r + 8*lh live across lanes 0..15) ----
        float nm[8];
#pragma unroll
        for (int r = 0; r < 8; ++r)
            nm[r] = fmaxf(fmaxf(s[0][r], s[1][r]), fmaxf(s[2][r], s[3][r]));
#pragma unroll
        for (int off = 1; off < 16; off <<= 1)
#pragma unroll
            for (int r = 0; r < 8; ++r)
                nm[r] = fmaxf(nm[r], __shfl_xor(nm[r], off, 32));

        float fac[8], ps[8];
#pragma unroll
        for (int r = 0; r < 8; ++r) {
            const float m2 = fmaxf(mrow[r], nm[r]);
            fac[r]  = __expf(mrow[r] - m2);
            mrow[r] = m2;
            ps[r]   = 0.0f;
        }
#pragma unroll
        for (int t = 0; t < 4; ++t)
#pragma unroll
            for (int r = 0; r < 8; ++r) {
                const float p = __expf(s[t][r] - mrow[r]);
                s[t][r] = p;
                ps[r]  += p;
            }
#pragma unroll
        for (int off = 1; off < 16; off <<= 1)
#pragma unroll
            for (int r = 0; r < 8; ++r)
                ps[r] += __shfl_xor(ps[r], off, 32);
#pragma unroll
        for (int r = 0; r < 8; ++r) lrow[r] = lrow[r] * fac[r] + ps[r];
#pragma unroll
        for (int t = 0; t < 8; ++t)
#pragma unroll
            for (int r = 0; r < 8; ++r) O[t][r] *= fac[r];

        // ---- P (D layout) -> LDS [M][n] so it can be re-read in A layout ----
#pragma unroll
        for (int t = 0; t < 4; ++t)
#pragma unroll
            for (int r = 0; r < 8; ++r) {
                const int M = r + 8 * lh;
                pw[M * 64 + 16 * t + l16] = f2bf(s[t][r]);
            }
        // wave-private region: same-wave LDS ops are in order, no barrier.

#if ATTN_USE_TDM
        // ---- drain the TDM, then make V visible to all waves -------------
        if (wid == 0) __builtin_amdgcn_s_wait_tensorcnt(0);
        __syncthreads();
#endif

        // ---- O += P . V'  (2 K-chunks of 32 x 8 channel tiles) ----
#pragma unroll
        for (int ck = 0; ck < 2; ++ck) {
            BF16V aP;   // A layout read back: two contiguous 16B runs per lane
            aP.q[0] = *(const uint4*)(pw + l16 * 64 + 32 * ck + 16 * lh);
            aP.q[1] = *(const uint4*)(pw + l16 * 64 + 32 * ck + 16 * lh + 8);
#pragma unroll
            for (int ct = 0; ct < 8; ++ct) {
                BF16V bV;  // lane col c = ch*128+16*ct+l16; element e -> n = 32ck+16lh+e
                const uint4* p = (const uint4*)(v_lds + (ch * 128 + 16 * ct + l16) * 64
                                                + 32 * ck + 16 * lh);
                bV.q[0] = p[0];
                bV.q[1] = p[1];
                O[ct] = __builtin_amdgcn_wmma_f32_16x16x32_bf16(
                    false, aP.v, false, bV.v, (short)0, O[ct], false, false);
            }
        }
    }

    // ---- epilogue: out = gamma * O / l + x (transpose scatter) ----
    const float g = gamma[0];
#pragma unroll
    for (int ct = 0; ct < 8; ++ct) {
        const int c = ch * 128 + 16 * ct + l16;
#pragma unroll
        for (int r = 0; r < 8; ++r) {
            const int M = r + 8 * lh;
            const int m = i0 + 16 * rg + M;
            const size_t idx = ((size_t)b * CC + c) * NN + m;
            out[idx] = g * (O[ct][r] / lrow[r]) + x[idx];
        }
    }
}

// ---------------------------------------------------------------------------
extern "C" void kernel_launch(void* const* d_in, const int* in_sizes, int n_in,
                              void* d_out, int out_size, void* d_ws, size_t ws_size,
                              hipStream_t stream) {
    const float* x     = (const float*)d_in[0];
    const float* Wq    = (const float*)d_in[1];
    const float* bq    = (const float*)d_in[2];
    const float* Wk    = (const float*)d_in[3];
    const float* bk    = (const float*)d_in[4];
    const float* Wv    = (const float*)d_in[5];
    const float* bv    = (const float*)d_in[6];
    const float* gamma = (const float*)d_in[7];
    float* out = (float*)d_out;

    // workspace: bf16 q, k, v  (10.5 MB total)
    unsigned short* qb = (unsigned short*)d_ws;
    unsigned short* kb = qb + (size_t)BB * CQKD * NN;
    unsigned short* vb = kb + (size_t)BB * CQKD * NN;

    const int nblk = BB * (NN / 64);
    proj_kernel<<<nblk, 256, 0, stream>>>(x, Wq, bq, Wk, bk, Wv, bv, qb, kb, vb);
    attn_kernel<<<nblk, 256, 0, stream>>>(qb, kb, vb, x, gamma, out);
}
